// Particle_17446157157101
// MI455X (gfx1250) — matrix operand order repaired and checked by reading, not compiled, not run on hardware
//
#include <hip/hip_runtime.h>

typedef __attribute__((ext_vector_type(16))) _Float16 v16h;
typedef __attribute__((ext_vector_type(8)))  _Float16 v8h;
typedef __attribute__((ext_vector_type(8)))  float    v8f;

#define N_PART   100000
#define N_EDGE   1600000
#define SDIM     64

__device__ __forceinline__ v8f wmma_f16(v16h a, v16h b, v8f c) {
  // D = A(16x32 f16) * B(32x16 f16) + C(16x16 f32)
  return __builtin_amdgcn_wmma_f32_16x16x32_f16(false, a, false, b, (short)0, c,
                                                false, false);
}

// A-fragment (16-bit A 16x32 layout, ISA 7.12.2): lane-half h holds
// K = k0..k0+7 (lo) and K = k0+16..k0+23 (hi), k0 = kbase + h*8.
__device__ __forceinline__ v16h a_frag(const _Float16* row, int k0) {
  v8h lo = *(const v8h*)(row + k0);
  v8h hi = *(const v8h*)(row + k0 + 16);
  return __builtin_shufflevector(lo, hi, 0, 1, 2, 3, 4, 5, 6, 7,
                                 8, 9, 10, 11, 12, 13, 14, 15);
}

// ---------------------------------------------------------------------------
// Pack W1(128x32), W2(32x32), W3(32x64) f32 row-major -> f16 WMMA B fragments.
// B 32x16 layout (16-bit B): lane l<16 holds column n=l, K=0..7,16..23;
// lane l+16 holds column n=l, K=8..15,24..31. Fragment f stored as
// 32 lanes x 16 contiguous halves (32B per lane).
// Frags: f=0..7 -> W1 (c = f>>1 K-chunk, t = f&1 N-tile); f=8,9 -> W2 tiles;
// f=10..13 -> W3 tiles.
// ---------------------------------------------------------------------------
__global__ void pack_wfrags(const float* __restrict__ W1,
                            const float* __restrict__ W2,
                            const float* __restrict__ W3,
                            _Float16* __restrict__ out) {
  int tid = blockIdx.x * blockDim.x + threadIdx.x;
  if (tid >= 14 * 512) return;
  int f    = tid >> 9;
  int idx  = tid & 511;
  int lane = idx >> 4;
  int q    = idx & 15;
  int l  = lane & 15;
  int hh = lane >> 4;
  int kk = (q < 8) ? hh * 8 + q : 16 + hh * 8 + (q - 8);  // K within the 32-wide chunk
  const float* W; int c, t, N;
  if (f < 8)       { W = W1; c = f >> 1; t = f & 1;  N = 32; }
  else if (f < 10) { W = W2; c = 0;      t = f - 8;  N = 32; }
  else             { W = W3; c = 0;      t = f - 10; N = 64; }
  int k = c * 32 + kk;
  int n = t * 16 + l;
  out[tid] = (_Float16)W[k * N + n];
}

__global__ void zero_f32(float* __restrict__ p, long n) {
  long i = (long)blockIdx.x * blockDim.x + threadIdx.x;
  if (i < n) p[i] = 0.f;
}

// ---------------------------------------------------------------------------
// 16-row 3-layer MLP: [16x128] -> relu[16x32] -> relu[16x32] -> [16x64]
// A   : this wave's 16x128 f16 input tile in LDS (row-major, stride 128)
// H   : this wave's 16x32 f16 scratch tile in LDS (row-major, stride 32)
// bw  : 14 B-fragments in registers. b1,b2: f32 biases.
// Output: 4 C-tiles (16x16 f32 each), bias3 NOT applied.
// Same-wave DS ops are in-order, so the load-old/store-new relayout through H
// needs no barriers (each wave owns its LDS slice).
// ---------------------------------------------------------------------------
struct Out4 { v8f t0, t1, t2, t3; };

__device__ __forceinline__ Out4 mlp3_16rows(const _Float16* __restrict__ A,
                                            _Float16* __restrict__ H,
                                            const v16h* bw,
                                            const float* __restrict__ b1,
                                            const float* __restrict__ b2,
                                            int lane) {
  const int l  = lane & 15;
  const int hh = lane >> 4;
  const v8f zero = {};

  // ---- layer 1: [16x128] @ [128x32] ----
  const _Float16* Ar = A + l * 128;
  v8f a0 = {}, a1 = {};
#pragma unroll
  for (int c = 0; c < 4; ++c) {
    v16h a = a_frag(Ar, c * 32 + hh * 8);
    a0 = wmma_f16(a, bw[2 * c + 0], a0);
    a1 = wmma_f16(a, bw[2 * c + 1], a1);
  }
  // bias+relu, relayout C (lane holds N=l, M=hh*8+j) -> row-major f16 in H
  float bb0 = b1[l], bb1 = b1[16 + l];
#pragma unroll
  for (int j = 0; j < 8; ++j) {
    int m = hh * 8 + j;
    H[m * 32 + l]      = (_Float16)fmaxf(a0[j] + bb0, 0.f);
    H[m * 32 + 16 + l] = (_Float16)fmaxf(a1[j] + bb1, 0.f);
  }

  // ---- layer 2: [16x32] @ [32x32] ----
  const _Float16* Hr = H + l * 32;
  v16h a2 = a_frag(Hr, hh * 8);
  v8f c0 = wmma_f16(a2, bw[8], zero);
  v8f c1 = wmma_f16(a2, bw[9], zero);
  float cb0 = b2[l], cb1 = b2[16 + l];
#pragma unroll
  for (int j = 0; j < 8; ++j) {
    int m = hh * 8 + j;
    H[m * 32 + l]      = (_Float16)fmaxf(c0[j] + cb0, 0.f);
    H[m * 32 + 16 + l] = (_Float16)fmaxf(c1[j] + cb1, 0.f);
  }

  // ---- layer 3: [16x32] @ [32x64] ----
  v16h a3 = a_frag(Hr, hh * 8);
  Out4 o;
  o.t0 = wmma_f16(a3, bw[10], zero);
  o.t1 = wmma_f16(a3, bw[11], zero);
  o.t2 = wmma_f16(a3, bw[12], zero);
  o.t3 = wmma_f16(a3, bw[13], zero);
  return o;
}

// ---------------------------------------------------------------------------
// Edge kernel: 8 waves/block, 16 edges/wave. Gather [x_src || x_dst] -> LDS,
// run MLP on WMMA, scatter-add 16x64 messages with hardware f32 atomics.
// ---------------------------------------------------------------------------
__global__ __launch_bounds__(256) void edge_kernel(
    const float* __restrict__ x,
    const int* __restrict__ esrc, const int* __restrict__ edst,
    const _Float16* __restrict__ wf,
    const float* __restrict__ b1, const float* __restrict__ b2,
    const float* __restrict__ b3,
    float* __restrict__ msg) {
  __shared__ _Float16 sA[8][16 * 128];
  __shared__ _Float16 sH[8][16 * 32];
  const int wave = threadIdx.x >> 5;
  const int lane = threadIdx.x & 31;
  const int e0 = (blockIdx.x * 8 + wave) * 16;

  // loop-invariant weight fragments -> registers (32B/lane/frag, L1-resident)
  v16h bw[14];
#pragma unroll
  for (int f = 0; f < 14; ++f)
    bw[f] = *(const v16h*)(wf + (f * 32 + lane) * 16);

  // gather phase: 2 lanes per edge-row; half 0 = x[src], half 1 = x[dst]
  {
    int r    = lane >> 1;
    int half = lane & 1;
    int node = half ? edst[e0 + r] : esrc[e0 + r];
    const float4* xp = (const float4*)(x + (long)node * SDIM);
    _Float16* dp = &sA[wave][r * 128 + half * 64];
#pragma unroll
    for (int i = 0; i < 16; ++i) {
      float4 v = xp[i];
      dp[i * 4 + 0] = (_Float16)v.x;
      dp[i * 4 + 1] = (_Float16)v.y;
      dp[i * 4 + 2] = (_Float16)v.z;
      dp[i * 4 + 3] = (_Float16)v.w;
    }
  }

  Out4 o = mlp3_16rows(sA[wave], sH[wave], bw, b1, b2, lane);

  const int l = lane & 15, hh = lane >> 4;
  float ob0 = b3[l], ob1 = b3[16 + l], ob2 = b3[32 + l], ob3 = b3[48 + l];
#pragma unroll
  for (int j = 0; j < 8; ++j) {
    int m = hh * 8 + j;
    float* mp = msg + (long)edst[e0 + m] * 64;
    unsafeAtomicAdd(mp + l,      o.t0[j] + ob0);
    unsafeAtomicAdd(mp + 16 + l, o.t1[j] + ob1);
    unsafeAtomicAdd(mp + 32 + l, o.t2[j] + ob2);
    unsafeAtomicAdd(mp + 48 + l, o.t3[j] + ob3);
  }
}

// ---------------------------------------------------------------------------
// Node kernel: 16 consecutive nodes/wave, input [x_i || msg_i], direct store.
// ---------------------------------------------------------------------------
__global__ __launch_bounds__(256) void node_kernel(
    const float* __restrict__ x, const float* __restrict__ msg,
    const _Float16* __restrict__ wf,
    const float* __restrict__ b1, const float* __restrict__ b2,
    const float* __restrict__ b3,
    float* __restrict__ out) {
  __shared__ _Float16 sA[8][16 * 128];
  __shared__ _Float16 sH[8][16 * 32];
  const int wave = threadIdx.x >> 5;
  const int lane = threadIdx.x & 31;
  const int tile = blockIdx.x * 8 + wave;
  if (tile * 16 >= N_PART) return;        // wave-uniform guard: EXEC stays ~0
  const int n0 = tile * 16;

  v16h bw[14];
#pragma unroll
  for (int f = 0; f < 14; ++f)
    bw[f] = *(const v16h*)(wf + (f * 32 + lane) * 16);

  {
    int r    = lane >> 1;
    int half = lane & 1;
    const float* srcp = (half ? msg : x) + (long)(n0 + r) * SDIM;
    const float4* xp = (const float4*)srcp;
    _Float16* dp = &sA[wave][r * 128 + half * 64];
#pragma unroll
    for (int i = 0; i < 16; ++i) {
      float4 v = xp[i];
      dp[i * 4 + 0] = (_Float16)v.x;
      dp[i * 4 + 1] = (_Float16)v.y;
      dp[i * 4 + 2] = (_Float16)v.z;
      dp[i * 4 + 3] = (_Float16)v.w;
    }
  }

  Out4 o = mlp3_16rows(sA[wave], sH[wave], bw, b1, b2, lane);

  const int l = lane & 15, hh = lane >> 4;
  float ob0 = b3[l], ob1 = b3[16 + l], ob2 = b3[32 + l], ob3 = b3[48 + l];
#pragma unroll
  for (int j = 0; j < 8; ++j) {
    int m = hh * 8 + j;
    float* op = out + (long)(n0 + m) * 64;
    op[l]      = o.t0[j] + ob0;
    op[16 + l] = o.t1[j] + ob1;
    op[32 + l] = o.t2[j] + ob2;
    op[48 + l] = o.t3[j] + ob3;
  }
}

extern "C" void kernel_launch(void* const* d_in, const int* in_sizes, int n_in,
                              void* d_out, int out_size, void* d_ws, size_t ws_size,
                              hipStream_t stream) {
  (void)in_sizes; (void)n_in; (void)out_size; (void)ws_size;
  const float* x    = (const float*)d_in[0];
  const int*   eidx = (const int*)d_in[1];      // int32 (JAX x64 disabled), [2, E]
  const float* mW1 = (const float*)d_in[2];
  const float* mb1 = (const float*)d_in[3];
  const float* mW2 = (const float*)d_in[4];
  const float* mb2 = (const float*)d_in[5];
  const float* mW3 = (const float*)d_in[6];
  const float* mb3 = (const float*)d_in[7];
  const float* oW1 = (const float*)d_in[8];
  const float* ob1 = (const float*)d_in[9];
  const float* oW2 = (const float*)d_in[10];
  const float* ob2 = (const float*)d_in[11];
  const float* oW3 = (const float*)d_in[12];
  const float* ob3 = (const float*)d_in[13];
  float* out = (float*)d_out;

  // workspace layout: [msg accumulator 25.6MB][edge wfrags 14KB][node wfrags 14KB]
  char* ws = (char*)d_ws;
  float*    msg = (float*)ws;
  _Float16* wfE = (_Float16*)(ws + (size_t)N_PART * 64 * sizeof(float));
  _Float16* wfN = wfE + 14 * 512;

  pack_wfrags<<<(14 * 512 + 255) / 256, 256, 0, stream>>>(mW1, mW2, mW3, wfE);
  pack_wfrags<<<(14 * 512 + 255) / 256, 256, 0, stream>>>(oW1, oW2, oW3, wfN);

  long nz = (long)N_PART * 64;
  zero_f32<<<(int)((nz + 255) / 256), 256, 0, stream>>>(msg, nz);

  // 1.6M edges / (8 waves * 16 edges) = 12500 blocks, exact
  edge_kernel<<<N_EDGE / 128, 256, 0, stream>>>(x, eidx, eidx + N_EDGE, wfE,
                                                mb1, mb2, mb3, msg);
  // 100000 nodes / 128 per block -> 782 blocks with wave-uniform guard
  node_kernel<<<(N_PART + 127) / 128, 256, 0, stream>>>(x, msg, wfN,
                                                        ob1, ob2, ob3, out);
}
